// CausalAttention_41961830482392
// MI455X (gfx1250) — matrix operand order repaired and checked by reading, not compiled, not run on hardware
//
#include <hip/hip_runtime.h>
#include <hip/hip_bf16.h>

#define N_TOK 8192
#define DIMSZ 512
#define INV_SQRT_D 0.04419417382415922f   // 1/sqrt(512)

typedef __bf16 bf16_t;
typedef __attribute__((ext_vector_type(16))) __bf16 v16bf;
typedef __attribute__((ext_vector_type(8)))  __bf16 v8bf;
typedef __attribute__((ext_vector_type(8)))  float  v8f;

typedef __attribute__((address_space(3))) char lds_char_t;

static __device__ __forceinline__ unsigned lds_addr_of(void* p) {
    return (unsigned)(unsigned long long)(lds_char_t*)p;
}

// Async 16B copy global -> LDS (per-lane), tracked by ASYNCcnt.
static __device__ __forceinline__ void async_g2l_b128(unsigned lds_off, const bf16_t* gptr) {
    asm volatile("global_load_async_to_lds_b128 %0, %1, off"
                 :: "v"(lds_off), "v"(gptr) : "memory");
}
static __device__ __forceinline__ void wait_async0() {
    asm volatile("s_wait_asynccnt 0x0" ::: "memory");
}

static __device__ __forceinline__ v8f wmma_bf16(v16bf a, v16bf b, v8f c) {
    // D = A(16x32 bf16) * B(32x16 bf16) + C(16x16 f32)
    return __builtin_amdgcn_wmma_f32_16x16x32_bf16(
        false, a, false, b, (short)0, c, false, false);
}

// A fragment: 16x32 bf16 tile, row-major source with leading dim ld.
// lane L holds row (L&15); elems [half*8 .. +7] and [16+half*8 .. +7], half=L>>4.
static __device__ __forceinline__ v16bf load_a_frag(const bf16_t* p, int ld, int lane) {
    const bf16_t* base = p + (size_t)(lane & 15) * ld + ((lane >> 4) << 3);
    v8bf lo = *(const v8bf*)base;
    v8bf hi = *(const v8bf*)(base + 16);
    v16bf out;
#pragma unroll
    for (int i = 0; i < 8; ++i) { out[i] = lo[i]; out[i + 8] = hi[i]; }
    return out;
}

// B fragment: B[k][n] = p[n*ld + k] (p holds the 16 "columns" of B as rows).
// lane L holds column (L&15); elems [half*8 .. +7] and [16+half*8 .. +7], half=L>>4.
static __device__ __forceinline__ v16bf load_b_frag(const bf16_t* p, int ld, int lane) {
    const bf16_t* base = p + (size_t)(lane & 15) * ld + ((lane >> 4) << 3);
    v8bf lo = *(const v8bf*)base;
    v8bf hi = *(const v8bf*)(base + 16);
    v16bf out;
#pragma unroll
    for (int i = 0; i < 8; ++i) { out[i] = lo[i]; out[i + 8] = hi[i]; }
    return out;
}

// ---------------------------------------------------------------- prep
__global__ void k_f32_to_bf16(const float* __restrict__ src, bf16_t* __restrict__ dst, int n) {
    int i = blockIdx.x * blockDim.x + threadIdx.x;
    if (i < n) dst[i] = (bf16_t)src[i];
}

__global__ void k_row_norm2(const float* __restrict__ T, float* __restrict__ tn2) {
    __shared__ float red[256];
    int row = blockIdx.x;
    const float* p = T + (size_t)row * DIMSZ;
    float s = 0.f;
    for (int k = threadIdx.x; k < DIMSZ; k += blockDim.x) { float v = p[k]; s += v * v; }
    red[threadIdx.x] = s;
    __syncthreads();
    for (int off = 128; off > 0; off >>= 1) {
        if ((int)threadIdx.x < off) red[threadIdx.x] += red[threadIdx.x + off];
        __syncthreads();
    }
    if (threadIdx.x == 0) tn2[row] = red[0];
}

// ---------------------------------------------------------------- GEMM: out = A @ W^T + bias
// mode 0: bf16 row-major; mode 1: bf16 transposed (out[col*M+row]); mode 2: f32 row-major
__global__ void k_gemm(const bf16_t* __restrict__ A, const bf16_t* __restrict__ W,
                       const float* __restrict__ bias, void* __restrict__ out,
                       int M, int Nn, int K, int mode) {
    int lane = threadIdx.x & 31;
    int w    = threadIdx.x >> 5;
    int row0 = blockIdx.y * 128 + w * 16;
    int col0 = blockIdx.x * 64;
    v8f acc[4] = {};
    for (int kk = 0; kk < K; kk += 32) {
        v16bf a = load_a_frag(A + (size_t)row0 * K + kk, K, lane);
#pragma unroll
        for (int t = 0; t < 4; ++t) {
            v16bf b = load_b_frag(W + (size_t)(col0 + 16 * t) * K + kk, K, lane);
            acc[t] = wmma_bf16(a, b, acc[t]);
        }
    }
    int half = lane >> 4, nl = lane & 15;
#pragma unroll
    for (int t = 0; t < 4; ++t) {
        int col = col0 + 16 * t + nl;
        float bv = bias ? bias[col] : 0.f;
#pragma unroll
        for (int r = 0; r < 8; ++r) {
            int row = row0 + r + 8 * half;
            float v = acc[t][r] + bv;
            if (mode == 0)      ((bf16_t*)out)[(size_t)row * Nn + col] = (bf16_t)v;
            else if (mode == 1) ((bf16_t*)out)[(size_t)col * M + row] = (bf16_t)v;
            else                ((float*)out)[(size_t)row * Nn + col] = v;
        }
    }
}

// ---------------------------------------------------------------- scale[i] = 1/(1+mean_j dist(i,j))
__global__ void k_scale(const bf16_t* __restrict__ Tbf, const float* __restrict__ tn2,
                        float* __restrict__ scal) {
    extern __shared__ bf16_t sT[];                 // 128 x 512 bf16 (128KB dynamic)
    int lane = threadIdx.x & 31;
    int w    = threadIdx.x >> 5;
    int rowBlock = blockIdx.x * 128;
    { // async stage of this block's 128 T rows (ASYNCcnt path, no VGPR transit)
        unsigned lbase = lds_addr_of((void*)sT);
        const bf16_t* gsrc = Tbf + (size_t)rowBlock * DIMSZ;
        for (int i = threadIdx.x; i < 128 * DIMSZ / 8; i += blockDim.x)
            async_g2l_b128(lbase + (unsigned)i * 16u, gsrc + (size_t)i * 8);
        wait_async0();
    }
    __syncthreads();
    int row0 = w * 16;
    int half = lane >> 4, nl = lane & 15;
    float partial[8], myn2[8];
#pragma unroll
    for (int r = 0; r < 8; ++r) {
        partial[r] = 0.f;
        myn2[r] = tn2[rowBlock + row0 + r + 8 * half];
    }
    for (int j = 0; j < N_TOK; j += 32) {
        if (j + 32 < N_TOK)
            __builtin_prefetch(Tbf + (size_t)(j + 32 + lane) * DIMSZ, 0, 3);
        v8f a0 = {}, a1 = {};
        for (int kk = 0; kk < DIMSZ; kk += 32) {
            v16bf a  = load_a_frag(sT + (size_t)row0 * DIMSZ + kk, DIMSZ, lane);
            v16bf b0 = load_b_frag(Tbf + (size_t)j * DIMSZ + kk, DIMSZ, lane);
            v16bf b1 = load_b_frag(Tbf + (size_t)(j + 16) * DIMSZ + kk, DIMSZ, lane);
            a0 = wmma_bf16(a, b0, a0);
            a1 = wmma_bf16(a, b1, a1);
        }
        float cn0 = tn2[j + nl], cn1 = tn2[j + 16 + nl];
#pragma unroll
        for (int r = 0; r < 8; ++r) {
            float d0 = myn2[r] + cn0 - 2.f * a0[r];
            float d1 = myn2[r] + cn1 - 2.f * a1[r];
            partial[r] += __fsqrt_rn(fmaxf(d0, 0.f) + 1e-8f);
            partial[r] += __fsqrt_rn(fmaxf(d1, 0.f) + 1e-8f);
        }
    }
#pragma unroll
    for (int mask = 1; mask <= 8; mask <<= 1)
#pragma unroll
        for (int r = 0; r < 8; ++r) partial[r] += __shfl_xor(partial[r], mask, 32);
    if (nl == 0) {
#pragma unroll
        for (int r = 0; r < 8; ++r) {
            int row = rowBlock + row0 + r + 8 * half;
            scal[row] = 1.f / (1.f + partial[r] * (1.f / (float)N_TOK));
        }
    }
}

// ---------------------------------------------------------------- pass1: per-row max & sum(exp)
__global__ void k_stats(const bf16_t* __restrict__ Qbf, const bf16_t* __restrict__ Kbf,
                        float* __restrict__ rowmax, float* __restrict__ rowsum) {
    extern __shared__ bf16_t sQ[];                 // 128 x 512 bf16
    int lane = threadIdx.x & 31;
    int w    = threadIdx.x >> 5;
    int rowBlock = blockIdx.x * 128;
    {
        unsigned lbase = lds_addr_of((void*)sQ);
        const bf16_t* gsrc = Qbf + (size_t)rowBlock * DIMSZ;
        for (int i = threadIdx.x; i < 128 * DIMSZ / 8; i += blockDim.x)
            async_g2l_b128(lbase + (unsigned)i * 16u, gsrc + (size_t)i * 8);
        wait_async0();
    }
    __syncthreads();
    int row0 = w * 16;
    int half = lane >> 4, nl = lane & 15;
    float m[8], l[8];
#pragma unroll
    for (int r = 0; r < 8; ++r) { m[r] = -3.0e38f; l[r] = 0.f; }
    for (int j = 0; j < N_TOK; j += 32) {
        if (j + 32 < N_TOK)
            __builtin_prefetch(Kbf + (size_t)(j + 32 + lane) * DIMSZ, 0, 3);
        v8f a0 = {}, a1 = {};
        for (int kk = 0; kk < DIMSZ; kk += 32) {
            v16bf a  = load_a_frag(sQ + (size_t)row0 * DIMSZ + kk, DIMSZ, lane);
            v16bf b0 = load_b_frag(Kbf + (size_t)j * DIMSZ + kk, DIMSZ, lane);
            v16bf b1 = load_b_frag(Kbf + (size_t)(j + 16) * DIMSZ + kk, DIMSZ, lane);
            a0 = wmma_bf16(a, b0, a0);
            a1 = wmma_bf16(a, b1, a1);
        }
#pragma unroll
        for (int r = 0; r < 8; ++r) {
            float v0 = a0[r] * INV_SQRT_D;
            float v1 = a1[r] * INV_SQRT_D;
            float mn = fmaxf(m[r], fmaxf(v0, v1));
            l[r] = l[r] * __expf(m[r] - mn) + __expf(v0 - mn) + __expf(v1 - mn);
            m[r] = mn;
        }
    }
#pragma unroll
    for (int mask = 1; mask <= 8; mask <<= 1) {
#pragma unroll
        for (int r = 0; r < 8; ++r) {
            float m2 = __shfl_xor(m[r], mask, 32);
            float l2 = __shfl_xor(l[r], mask, 32);
            float mn = fmaxf(m[r], m2);
            l[r] = l[r] * __expf(m[r] - mn) + l2 * __expf(m2 - mn);
            m[r] = mn;
        }
    }
    if (nl == 0) {
#pragma unroll
        for (int r = 0; r < 8; ++r) {
            int row = rowBlock + row0 + r + 8 * half;
            rowmax[row] = m[r];
            rowsum[row] = l[r];
        }
    }
}

// ---------------------------------------------------------------- pass2: O = (softmax * scale) @ V
// block = 256 threads = 8 waves. 2 groups of 4 waves; group g owns 16 Q rows.
// Within a group the 4 waves split the K-reduction of S (no redundant QK^T),
// reduce in LDS, exp -> P (bf16, LDS), then each wave does P @ V[:, q*128 .. +127].
// Q A-fragments are loop-invariant per wave -> hoisted into 32 VGPRs.
__global__ void k_attn_av(const bf16_t* __restrict__ Qbf, const bf16_t* __restrict__ Kbf,
                          const bf16_t* __restrict__ Vt, const float* __restrict__ rowmax,
                          const float* __restrict__ rowsum, const float* __restrict__ scal,
                          bf16_t* __restrict__ Obf) {
    __shared__ float  sS[2 * 4 * 2 * 256]; // 16KB partial S: [g][q][tile][lane][r]
    __shared__ bf16_t sPb[2 * 16 * 32];    // 2KB  P tiles
    __shared__ float  sM[32], sC[32];

    int lane = threadIdx.x & 31;
    int w    = threadIdx.x >> 5;
    int g    = w >> 2;          // group (0/1): which 16 rows
    int q    = w & 3;           // wave-in-group: k-slice for S, col-slice for O
    int rowBase = blockIdx.x * 32;
    int rowG = rowBase + g * 16;
    int half = lane >> 4, nl = lane & 15;

    if (threadIdx.x < 32) {
        int row = rowBase + threadIdx.x;
        sM[threadIdx.x] = rowmax[row];
        sC[threadIdx.x] = scal[row] / rowsum[row];
    }
    __syncthreads();

    // hoist this wave's 4 Q A-fragments (rows rowG..rowG+15, k = q*128 .. +127)
    v16bf qa[4];
#pragma unroll
    for (int ks = 0; ks < 4; ++ks)
        qa[ks] = load_a_frag(Qbf + (size_t)rowG * DIMSZ + q * 128 + ks * 32, DIMSZ, lane);

    v8f accO[8] = {};
    int tgi = threadIdx.x & 127;   // thread index within group

    for (int j = 0; j < N_TOK; j += 32) {
        if (j + 32 < N_TOK) {
            __builtin_prefetch(Kbf + (size_t)(j + 32 + lane) * DIMSZ, 0, 3);
            __builtin_prefetch(Vt + (size_t)(q * 128 + lane) * N_TOK + j + 32, 0, 3);
        }
        // ---- partial S over this wave's k-slice
        v8f a0 = {}, a1 = {};
        int kbase = q * 128;
#pragma unroll
        for (int ks = 0; ks < 4; ++ks) {
            int kk = kbase + ks * 32;
            v16bf b0 = load_b_frag(Kbf + (size_t)j * DIMSZ + kk, DIMSZ, lane);
            v16bf b1 = load_b_frag(Kbf + (size_t)(j + 16) * DIMSZ + kk, DIMSZ, lane);
            a0 = wmma_bf16(qa[ks], b0, a0);
            a1 = wmma_bf16(qa[ks], b1, a1);
        }
        *(v8f*)&sS[(((g * 4 + q) * 2) + 0) * 256 + lane * 8] = a0;
        *(v8f*)&sS[(((g * 4 + q) * 2) + 1) * 256 + lane * 8] = a1;
        __syncthreads();

        // ---- reduce partials, softmax-exp, emit P (bf16) to LDS
#pragma unroll
        for (int e = 0; e < 4; ++e) {
            int idx  = tgi * 4 + e;        // 0..511 of the 16x32 P tile
            int prow = idx >> 5;
            int pcol = idx & 31;
            int r    = prow & 7;
            int li   = (pcol & 15) | ((prow >> 3) << 4);
            int tile = pcol >> 4;
            float s = 0.f;
#pragma unroll
            for (int qq = 0; qq < 4; ++qq)
                s += sS[(((g * 4 + qq) * 2) + tile) * 256 + li * 8 + r];
            float v = s * INV_SQRT_D;
            float p = __expf(v - sM[g * 16 + prow]) * sC[g * 16 + prow];
            sPb[g * 512 + prow * 32 + pcol] = (bf16_t)p;
        }
        __syncthreads();

        // ---- O += P(16x32) @ V[j..j+31, q*128 .. +127]   (B frags from V^T: contiguous)
        v16bf pa = load_a_frag(&sPb[g * 512], 32, lane);
#pragma unroll
        for (int nt = 0; nt < 8; ++nt) {
            const bf16_t* bp = Vt + (size_t)(q * 128 + nt * 16) * N_TOK + j;
            v16bf b = load_b_frag(bp, N_TOK, lane);
            accO[nt] = wmma_bf16(pa, b, accO[nt]);
        }
    }

#pragma unroll
    for (int nt = 0; nt < 8; ++nt) {
        int col = q * 128 + nt * 16 + nl;
#pragma unroll
        for (int r = 0; r < 8; ++r) {
            int row = rowG + r + 8 * half;
            Obf[(size_t)row * DIMSZ + col] = (bf16_t)accO[nt][r];
        }
    }
}

// ----------------------------------------------------------------
extern "C" void kernel_launch(void* const* d_in, const int* in_sizes, int n_in,
                              void* d_out, int out_size, void* d_ws, size_t ws_size,
                              hipStream_t stream) {
    (void)in_sizes; (void)n_in; (void)out_size; (void)ws_size;
    const float* H  = (const float*)d_in[0];
    const float* T  = (const float*)d_in[1];
    const float* Wq = (const float*)d_in[2];
    const float* bq = (const float*)d_in[3];
    const float* Wk = (const float*)d_in[4];
    const float* bk = (const float*)d_in[5];
    const float* Wv = (const float*)d_in[6];
    const float* bv = (const float*)d_in[7];
    const float* Wo = (const float*)d_in[8];
    const float* bo = (const float*)d_in[9];

    char* ws = (char*)d_ws;
    size_t off = 0;
    auto take = [&](size_t bytes) {
        char* p = ws + off;
        off += (bytes + 255) & ~(size_t)255;
        return p;
    };
    const size_t HD = (size_t)N_TOK * DIMSZ;
    const size_t WW = (size_t)DIMSZ * DIMSZ;
    bf16_t* Hbf  = (bf16_t*)take(HD * 2);
    bf16_t* Tbf  = (bf16_t*)take(HD * 2);
    bf16_t* Qbf  = (bf16_t*)take(HD * 2);
    bf16_t* Kbf  = (bf16_t*)take(HD * 2);
    bf16_t* Vt   = (bf16_t*)take(HD * 2);   // V transposed: [DIMSZ][N_TOK]
    bf16_t* Obf  = (bf16_t*)take(HD * 2);
    bf16_t* Wqb  = (bf16_t*)take(WW * 2);
    bf16_t* Wkb  = (bf16_t*)take(WW * 2);
    bf16_t* Wvb  = (bf16_t*)take(WW * 2);
    bf16_t* Wob  = (bf16_t*)take(WW * 2);
    float*  tn2  = (float*)take((size_t)N_TOK * 4);
    float*  scal = (float*)take((size_t)N_TOK * 4);
    float*  rmx  = (float*)take((size_t)N_TOK * 4);
    float*  rsm  = (float*)take((size_t)N_TOK * 4);

    int nHD = (int)HD, nWW = (int)WW;
    k_f32_to_bf16<<<(nHD + 255) / 256, 256, 0, stream>>>(H, Hbf, nHD);
    k_f32_to_bf16<<<(nHD + 255) / 256, 256, 0, stream>>>(T, Tbf, nHD);
    k_f32_to_bf16<<<(nWW + 255) / 256, 256, 0, stream>>>(Wq, Wqb, nWW);
    k_f32_to_bf16<<<(nWW + 255) / 256, 256, 0, stream>>>(Wk, Wkb, nWW);
    k_f32_to_bf16<<<(nWW + 255) / 256, 256, 0, stream>>>(Wv, Wvb, nWW);
    k_f32_to_bf16<<<(nWW + 255) / 256, 256, 0, stream>>>(Wo, Wob, nWW);
    k_row_norm2<<<N_TOK, 256, 0, stream>>>(T, tn2);

    dim3 gg(DIMSZ / 64, N_TOK / 128);
    k_gemm<<<gg, 256, 0, stream>>>(Hbf, Wqb, bq, Qbf, N_TOK, DIMSZ, DIMSZ, 0);
    k_gemm<<<gg, 256, 0, stream>>>(Hbf, Wkb, bk, Kbf, N_TOK, DIMSZ, DIMSZ, 0);
    k_gemm<<<gg, 256, 0, stream>>>(Hbf, Wvb, bv, Vt,  N_TOK, DIMSZ, DIMSZ, 1);

    k_scale<<<N_TOK / 128, 256, 128 * 1024, stream>>>(Tbf, tn2, scal);
    k_stats<<<N_TOK / 128, 256, 128 * 1024, stream>>>(Qbf, Kbf, rmx, rsm);
    k_attn_av<<<N_TOK / 32, 256, 0, stream>>>(Qbf, Kbf, Vt, rmx, rsm, scal, Obf);

    k_gemm<<<gg, 256, 0, stream>>>(Obf, Wob, bo, d_out, N_TOK, DIMSZ, DIMSZ, 2);
}
